// MultiHeadedAttention_34668976013902
// MI455X (gfx1250) — compile-verified, matrix operand order checked by reading, not run on hardware
//
#include <hip/hip_runtime.h>

// ---------------------------------------------------------------------------
// MultiHeadedAttention for gfx1250 (MI455X): bf16 WMMA everywhere.
//   B=2, S=2048, D_MODEL=1024, H=16, d_k=64
// ---------------------------------------------------------------------------

typedef __attribute__((ext_vector_type(16))) __bf16 v16bf;
typedef __attribute__((ext_vector_type(8)))  __bf16 v8bf;
typedef __attribute__((ext_vector_type(8)))  float  v8f;

static constexpr int DM   = 1024;  // d_model
static constexpr int SEQ  = 2048;  // sequence length
static constexpr int NH   = 16;    // heads
static constexpr int DK   = 64;    // head dim
static constexpr int ROWS = 2 * SEQ;      // B*S = 4096 rows
static constexpr int KP32 = 40;    // LDS pitch (bf16 elems) for 32-wide tiles (80B, 16B-mult)
static constexpr int KP64 = 72;    // LDS pitch for 64-wide tiles (144B, 16B-mult)

__device__ __forceinline__ __bf16 f2bf(float f) {
  union { float f; unsigned u; } in; in.f = f;
  unsigned u = in.u;
  u += 0x7FFFu + ((u >> 16) & 1u);           // round-to-nearest-even
  union { unsigned short s; __bf16 b; } out; out.s = (unsigned short)(u >> 16);
  return out.b;
}
__device__ __forceinline__ __bf16 cvt_bf16(float x)  { return f2bf(x); }
__device__ __forceinline__ __bf16 cvt_bf16(__bf16 x) { return x; }

// output converters (declared BEFORE the template that uses them)
__device__ __forceinline__ float  cvt_bf16_out(float x, float*)  { return x; }
__device__ __forceinline__ __bf16 cvt_bf16_out(float x, __bf16*) { return f2bf(x); }

__device__ __forceinline__ v8f zero8() {
  v8f z;
#pragma unroll
  for (int i = 0; i < 8; ++i) z[i] = 0.0f;
  return z;
}

// Load one 16x32 bf16 A/B fragment (both use the same per-lane pattern):
// lane half 0 holds K = {0..7, 16..23}, half 1 holds K = {8..15, 24..31},
// i.e. two contiguous 16-byte chunks at +hf*8 and +16+hf*8 elements.
__device__ __forceinline__ v16bf load_frag(const __bf16* rowp, int hf) {
  union { v16bf v; v8bf h[2]; } u;
  u.h[0] = *(const v8bf*)(rowp + hf * 8);
  u.h[1] = *(const v8bf*)(rowp + 16 + hf * 8);
  return u.v;
}

__device__ __forceinline__ v8f wmma_bf16(v16bf a, v16bf b, v8f c) {
  return __builtin_amdgcn_wmma_f32_16x16x32_bf16(false, a, false, b,
                                                 (short)0, c, false, false);
}

// ---------------------------------------------------------------------------
// GEMM: out[m][n] = sum_k A[m][k] * W[n][k] + bias[n]   (torch Linear)
// A: ROWS x DM (float or bf16), W: DM x DM float, out: ROWS x DM (OT)
// Block = 128 threads = 4 waves. Workgroup tile 64(M) x 64(N), K-step 32.
// Wave w computes rows [w*16, w*16+16) of the tile across all 64 N cols.
// ---------------------------------------------------------------------------
template <typename AT, typename OT>
__global__ __launch_bounds__(128) void gemm_wmma(const AT* __restrict__ A,
                                                 const float* __restrict__ W,
                                                 const float* __restrict__ bias,
                                                 OT* __restrict__ out) {
  __shared__ __bf16 lA[64 * KP32];
  __shared__ __bf16 lB[64 * KP32];

  const int m0 = blockIdx.x * 64;
  const int n0 = blockIdx.y * 64;
  const int t  = threadIdx.x;
  const int wv = t >> 5;
  const int lane = t & 31;
  const int l16 = lane & 15;
  const int hf  = lane >> 4;

  v8f acc[4];
#pragma unroll
  for (int i = 0; i < 4; ++i) acc[i] = zero8();

  const int srow = t >> 1;          // 0..63, staging row
  const int skb  = (t & 1) * 16;    // 0 or 16, staging k-chunk

  for (int k0 = 0; k0 < DM; k0 += 32) {
    {  // stage A tile (64 x 32), convert to bf16
      const AT* src = A + (size_t)(m0 + srow) * DM + k0 + skb;
      __bf16* dst = &lA[srow * KP32 + skb];
#pragma unroll
      for (int i = 0; i < 16; ++i) dst[i] = cvt_bf16(src[i]);
    }
    {  // stage W tile (64 rows of W = 64 output cols)
      const float* src = W + (size_t)(n0 + srow) * DM + k0 + skb;
      __bf16* dst = &lB[srow * KP32 + skb];
#pragma unroll
      for (int i = 0; i < 16; ++i) dst[i] = f2bf(src[i]);
    }
    if (k0 + 32 < DM) {
      __builtin_prefetch(A + (size_t)(m0 + srow) * DM + k0 + 32 + skb, 0, 1);
      __builtin_prefetch(W + (size_t)(n0 + srow) * DM + k0 + 32 + skb, 0, 1);
    }
    __syncthreads();

    const v16bf a = load_frag(&lA[(wv * 16 + l16) * KP32], hf);
#pragma unroll
    for (int nt = 0; nt < 4; ++nt) {
      const v16bf b = load_frag(&lB[(nt * 16 + l16) * KP32], hf);
      acc[nt] = wmma_bf16(a, b, acc[nt]);
    }
    __syncthreads();
  }

  // Epilogue: C layout -> element (m = r + 8*hf, n = l16 + 16*nt)
#pragma unroll
  for (int nt = 0; nt < 4; ++nt) {
    const int ncol = n0 + nt * 16 + l16;
    const float bv = bias[ncol];
#pragma unroll
    for (int r = 0; r < 8; ++r) {
      const int m = m0 + wv * 16 + hf * 8 + r;
      out[(size_t)m * DM + ncol] = cvt_bf16_out(acc[nt][r] + bv, (OT*)nullptr);
    }
  }
}

// ---------------------------------------------------------------------------
// Flash attention per (b, h, 64-query block). 4 waves; wave owns 16 q-rows.
// Q,K,V: bf16 [B,S,DM] with head h occupying cols [h*64, h*64+64).
// ---------------------------------------------------------------------------
__global__ __launch_bounds__(128) void attn_wmma(const __bf16* __restrict__ Q,
                                                 const __bf16* __restrict__ Km,
                                                 const __bf16* __restrict__ Vm,
                                                 const int* __restrict__ mask,
                                                 __bf16* __restrict__ O) {
  __shared__ __bf16 lK[32 * KP64];       // K tile, [j][d] row-major
  __shared__ __bf16 lVt[64 * KP32];      // V tile transposed, [d][j]
  __shared__ __bf16 lP[4 * 16 * KP32];   // per-wave P scratch, [m][n]

  const int qb = blockIdx.x * 64;
  const int h  = blockIdx.y;
  const int b  = blockIdx.z;
  const int t  = threadIdx.x;
  const int wv = t >> 5;
  const int lane = t & 31;
  const int l16 = lane & 15;
  const int hf  = lane >> 4;

  const size_t headoff = (size_t)b * SEQ * DM + (size_t)h * DK;

  // Q fragments for this wave's 16 rows: d in [0,32) and [32,64)
  v16bf qf[2];
  {
    const __bf16* qp = Q + headoff + (size_t)(qb + wv * 16 + l16) * DM;
    qf[0] = load_frag(qp, hf);
    qf[1] = load_frag(qp + 32, hf);
  }

  float m_i[8], l_i[8];
#pragma unroll
  for (int r = 0; r < 8; ++r) { m_i[r] = -1e30f; l_i[r] = 0.0f; }
  v8f o[4];
#pragma unroll
  for (int i = 0; i < 4; ++i) o[i] = zero8();

  const int sj = t >> 2;          // 0..31 staging row (key index in tile)
  const int sd = (t & 3) * 16;    // 0,16,32,48 staging d-chunk

  for (int kt = 0; kt < SEQ; kt += 32) {
    {  // stage K tile row-major
      const __bf16* src = Km + headoff + (size_t)(kt + sj) * DM + sd;
      *(v8bf*)&lK[sj * KP64 + sd]     = *(const v8bf*)src;
      *(v8bf*)&lK[sj * KP64 + sd + 8] = *(const v8bf*)(src + 8);
      // stage V tile transposed: lVt[d][j]
      const __bf16* vsrc = Vm + headoff + (size_t)(kt + sj) * DM + sd;
#pragma unroll
      for (int i = 0; i < 16; ++i) lVt[(sd + i) * KP32 + sj] = vsrc[i];
    }
    if (kt + 32 < SEQ) {
      __builtin_prefetch(Km + headoff + (size_t)(kt + 32 + sj) * DM + sd, 0, 1);
      __builtin_prefetch(Vm + headoff + (size_t)(kt + 32 + sj) * DM + sd, 0, 1);
    }
    __syncthreads();

    // scores S[16][32] = Q(16x64) . K_tile^T(64x32)
    v8f s[2] = {zero8(), zero8()};
#pragma unroll
    for (int kk = 0; kk < 2; ++kk) {
#pragma unroll
      for (int nt = 0; nt < 2; ++nt) {
        const v16bf bfrag = load_frag(&lK[(nt * 16 + l16) * KP64 + kk * 32], hf);
        s[nt] = wmma_bf16(qf[kk], bfrag, s[nt]);
      }
    }

    // scale + mask + online softmax (row lives across 16-lane halves)
    const int qrow0 = qb + wv * 16 + hf * 8;
    const size_t mbase = (size_t)b * SEQ * SEQ;
#pragma unroll
    for (int r = 0; r < 8; ++r) {
      const int qr = qrow0 + r;
      float v0 = s[0][r] * 0.125f;  // 1/sqrt(64)
      float v1 = s[1][r] * 0.125f;
      const int mv0 = mask[mbase + (size_t)qr * SEQ + kt + l16];
      const int mv1 = mask[mbase + (size_t)qr * SEQ + kt + 16 + l16];
      v0 = mv0 ? v0 : -1e9f;
      v1 = mv1 ? v1 : -1e9f;
      float mx = fmaxf(v0, v1);
#pragma unroll
      for (int off = 8; off >= 1; off >>= 1)
        mx = fmaxf(mx, __shfl_xor(mx, off, 16));
      const float mnew = fmaxf(m_i[r], mx);
      const float p0 = __expf(v0 - mnew);
      const float p1 = __expf(v1 - mnew);
      float rs = p0 + p1;
#pragma unroll
      for (int off = 8; off >= 1; off >>= 1) rs += __shfl_xor(rs, off, 16);
      const float alpha = __expf(m_i[r] - mnew);
      l_i[r] = l_i[r] * alpha + rs;
      m_i[r] = mnew;
#pragma unroll
      for (int nt = 0; nt < 4; ++nt) o[nt][r] *= alpha;
      // spill P (C layout) to per-wave LDS for re-read in A-fragment layout
      __bf16* pp = &lP[(wv * 16 + hf * 8 + r) * KP32];
      pp[l16]      = f2bf(p0);
      pp[16 + l16] = f2bf(p1);
    }

    // O += P(16x32) . V_tile(32x64) ; B-fragments come from transposed V
    const v16bf pf = load_frag(&lP[(wv * 16 + l16) * KP32], hf);
#pragma unroll
    for (int nt = 0; nt < 4; ++nt) {
      const v16bf vf = load_frag(&lVt[(nt * 16 + l16) * KP32], hf);
      o[nt] = wmma_bf16(pf, vf, o[nt]);
    }
    __syncthreads();
  }

  // epilogue: O /= l, store bf16 into [B,S,DM] at head cols
#pragma unroll
  for (int r = 0; r < 8; ++r) {
    const float inv = 1.0f / l_i[r];
    const int qr = qb + wv * 16 + hf * 8 + r;
#pragma unroll
    for (int nt = 0; nt < 4; ++nt) {
      O[headoff + (size_t)qr * DM + nt * 16 + l16] = f2bf(o[nt][r] * inv);
    }
  }
}

// ---------------------------------------------------------------------------
// Host-side launcher
// ---------------------------------------------------------------------------
extern "C" void kernel_launch(void* const* d_in, const int* in_sizes, int n_in,
                              void* d_out, int out_size, void* d_ws, size_t ws_size,
                              hipStream_t stream) {
  const float* query = (const float*)d_in[0];
  const float* key   = (const float*)d_in[1];
  const float* value = (const float*)d_in[2];
  const int*   mask  = (const int*)d_in[3];
  const float* Wq = (const float*)d_in[4];
  const float* bq = (const float*)d_in[5];
  const float* Wk = (const float*)d_in[6];
  const float* bk = (const float*)d_in[7];
  const float* Wv = (const float*)d_in[8];
  const float* bv = (const float*)d_in[9];
  const float* Wo = (const float*)d_in[10];
  const float* bo = (const float*)d_in[11];
  float* out = (float*)d_out;

  const size_t NELEM = (size_t)ROWS * DM;  // 4096*1024
  __bf16* Qws = (__bf16*)d_ws;
  __bf16* Kws = Qws + NELEM;
  __bf16* Vws = Kws + NELEM;
  __bf16* Aws = Vws + NELEM;  // attention output (pre out-proj)

  dim3 gGemm(ROWS / 64, DM / 64);  // (64, 16)
  dim3 blk(128);

  gemm_wmma<float, __bf16><<<gGemm, blk, 0, stream>>>(query, Wq, bq, Qws);
  gemm_wmma<float, __bf16><<<gGemm, blk, 0, stream>>>(key,   Wk, bk, Kws);
  gemm_wmma<float, __bf16><<<gGemm, blk, 0, stream>>>(value, Wv, bv, Vws);

  dim3 gAttn(SEQ / 64, NH, 2);  // (32, 16, 2)
  attn_wmma<<<gAttn, blk, 0, stream>>>(Qws, Kws, Vws, mask, Aws);

  gemm_wmma<__bf16, float><<<gGemm, blk, 0, stream>>>(Aws, Wo, bo, out);
}
